// lossAnVall_con_39642548142296
// MI455X (gfx1250) — matrix operand order repaired and checked by reading, not compiled, not run on hardware
//
#include <hip/hip_runtime.h>

typedef __attribute__((ext_vector_type(16))) _Float16 v16h;
typedef __attribute__((ext_vector_type(8)))  _Float16 v8h;
typedef __attribute__((ext_vector_type(8)))  float    v8f;

static constexpr int Dk  = 256;   // feature dim
static constexpr int Mm  = 8192;  // B*N
static constexpr int BT  = 128;   // block tile (rows x cols of sim)

#define TEMP_F 0.07f
#define W00_F  0.1f
#define EPS_F  1e-8f

// ---------------------------------------------------------------------------
// Kernel 1: L2-normalize each row of a and v, store as f16 into workspace.
// One wave32 per row; lane holds 8 strided elements of the 256-wide row.
// ---------------------------------------------------------------------------
__global__ void __launch_bounds__(256)
normalize_kernel(const float* __restrict__ a,
                 const float* __restrict__ v,
                 _Float16* __restrict__ af,
                 _Float16* __restrict__ vf) {
    const int gw   = blockIdx.x * (blockDim.x >> 5) + (threadIdx.x >> 5);
    const int lane = threadIdx.x & 31;
    const float* src;
    _Float16*    dst;
    if (gw < Mm) { src = a + (size_t)gw * Dk;        dst = af + (size_t)gw * Dk; }
    else         { src = v + (size_t)(gw - Mm) * Dk; dst = vf + (size_t)(gw - Mm) * Dk; }

    float vals[8];
    float ss = 0.0f;
#pragma unroll
    for (int i = 0; i < 8; ++i) {
        vals[i] = src[lane + i * 32];
        ss += vals[i] * vals[i];
    }
#pragma unroll
    for (int m = 16; m >= 1; m >>= 1) ss += __shfl_xor(ss, m, 32);

    const float inv = 1.0f / fmaxf(sqrtf(ss), 1e-12f);
#pragma unroll
    for (int i = 0; i < 8; ++i)
        dst[lane + i * 32] = (_Float16)(vals[i] * inv);
}

// ---------------------------------------------------------------------------
// Kernel 2: zero the global accumulators (4*M floats).
// ---------------------------------------------------------------------------
__global__ void zero_kernel(float* __restrict__ p, int n) {
    int i = blockIdx.x * blockDim.x + threadIdx.x;
    if (i < n) p[i] = 0.0f;
}

// ---------------------------------------------------------------------------
// Kernel 3: fused  exp(sim/T) tile GEMM + masked row/col reductions.
// Block = 256 threads = 8 waves; block tile = 128x128 of sim.
// Wave (wm,wn): wm in 0..3, wn in 0..1 owns a 32x64 sub-tile:
//   2 (M) x 4 (N) accumulators of v_wmma_f32_16x16x32_f16, K-loop of 8.
// Epilogue reduces in registers (shuffles), then few conflict-free LDS adds.
// ---------------------------------------------------------------------------
__global__ void __launch_bounds__(256)
sim_tile_kernel(const _Float16* __restrict__ af,
                const _Float16* __restrict__ vf,
                const int* __restrict__ la,
                const int* __restrict__ lv,
                float* __restrict__ rowD, float* __restrict__ rowN,
                float* __restrict__ colD, float* __restrict__ colN) {
    __shared__ float sRowD[BT], sRowN[BT], sColD[BT], sColN[BT];
    __shared__ int   sLa[BT], sLv[BT];

    const int bm = blockIdx.y * BT;   // sim rows (audio)
    const int bn = blockIdx.x * BT;   // sim cols (visual)
    const int t  = threadIdx.x;

    if (t < BT) {
        sRowD[t] = 0.0f; sRowN[t] = 0.0f;
        sColD[t] = 0.0f; sColN[t] = 0.0f;
        sLa[t] = la[bm + t];
        sLv[t] = lv[bn + t];
    }
    __syncthreads();

    const int wave = t >> 5;
    const int lane = t & 31;
    const int wm   = wave >> 1;   // 0..3
    const int wn   = wave & 1;    // 0..1
    const int g    = lane >> 4;   // lane group (0/1)
    const int l15  = lane & 15;

    v8f acc[2][4] = {};

    const int rowA0 = bm + wm * 32 + l15;   // A row for this lane (per ti tile)
    const int colB0 = bn + wn * 64 + l15;   // B column for this lane (per tj tile)

    for (int k0 = 0; k0 < Dk; k0 += 32) {
        // A fragments: lane group g holds K = g*8+[0..7] (elems 0..7 -> VGPR0..3)
        // and K = 16+g*8+[0..7] (elems 8..15 -> VGPR4..7).
        v16h afrag[2];
#pragma unroll
        for (int ti = 0; ti < 2; ++ti) {
            const _Float16* p = af + (size_t)(rowA0 + ti * 16) * Dk + k0 + g * 8;
            v8h lo = *(const v8h*)(p);
            v8h hi = *(const v8h*)(p + 16);
            afrag[ti] = __builtin_shufflevector(lo, hi,
                0,1,2,3,4,5,6,7,8,9,10,11,12,13,14,15);
        }
        // B fragments: lane = column (l15), lane group g holds K = g*8+[0..7]
        // and K = 16+g*8+[0..7], matching A (sim = a . v^T).
        v16h bfrag[4];
#pragma unroll
        for (int tj = 0; tj < 4; ++tj) {
            const _Float16* p = vf + (size_t)(colB0 + tj * 16) * Dk + k0 + g * 8;
            v8h lo = *(const v8h*)(p);
            v8h hi = *(const v8h*)(p + 16);
            bfrag[tj] = __builtin_shufflevector(lo, hi,
                0,1,2,3,4,5,6,7,8,9,10,11,12,13,14,15);
        }
#pragma unroll
        for (int ti = 0; ti < 2; ++ti)
#pragma unroll
            for (int tj = 0; tj < 4; ++tj)
                acc[ti][tj] = __builtin_amdgcn_wmma_f32_16x16x32_f16(
                    false, afrag[ti], false, bfrag[tj],
                    (short)0, acc[ti][tj], false, false);
    }

    // ---- Epilogue -------------------------------------------------------
    // C/D layout: VGPR r, lane l -> M = r + (l>>4)*8, N = l&15.
    // Hoist labels into registers.
    int laR[2][8];
#pragma unroll
    for (int ti = 0; ti < 2; ++ti)
#pragma unroll
        for (int r = 0; r < 8; ++r)
            laR[ti][r] = sLa[wm * 32 + ti * 16 + g * 8 + r];
    int lvC[4];
#pragma unroll
    for (int tj = 0; tj < 4; ++tj)
        lvC[tj] = sLv[wn * 64 + tj * 16 + l15];

    const float invT = 1.0f / TEMP_F;
    float rowE[2][8] = {}, rowEW[2][8] = {};
    float colE[4]    = {}, colEW[4]    = {};

#pragma unroll
    for (int ti = 0; ti < 2; ++ti) {
#pragma unroll
        for (int tj = 0; tj < 4; ++tj) {
#pragma unroll
            for (int r = 0; r < 8; ++r) {
                const float e = __expf(acc[ti][tj][r] * invT);
                const int laa = laR[ti][r];
                const float w = (laa == lvC[tj]) ? (laa ? 1.0f : W00_F) : 0.0f;
                const float ew = e * w;
                rowE[ti][r]  += e;   rowEW[ti][r] += ew;
                colE[tj]     += e;   colEW[tj]    += ew;
            }
        }
    }

    // Columns: lanes l and l+16 hold the same column (different row halves).
#pragma unroll
    for (int tj = 0; tj < 4; ++tj) {
        colE[tj]  += __shfl_xor(colE[tj], 16, 32);
        colEW[tj] += __shfl_xor(colEW[tj], 16, 32);
    }
    if (lane < 16) {
#pragma unroll
        for (int tj = 0; tj < 4; ++tj) {
            const int nLoc = wn * 64 + tj * 16 + l15;
            unsafeAtomicAdd(&sColD[nLoc], colE[tj]);
            unsafeAtomicAdd(&sColN[nLoc], colEW[tj]);
        }
    }

    // Rows: butterfly across the 16 lanes of each half (masks 1,2,4,8 stay
    // inside the half), then lanes with l15==0 (lane 0 and lane 16) commit.
#pragma unroll
    for (int ti = 0; ti < 2; ++ti) {
#pragma unroll
        for (int r = 0; r < 8; ++r) {
#pragma unroll
            for (int m = 1; m <= 8; m <<= 1) {
                rowE[ti][r]  += __shfl_xor(rowE[ti][r],  m, 32);
                rowEW[ti][r] += __shfl_xor(rowEW[ti][r], m, 32);
            }
        }
    }
    if (l15 == 0) {
#pragma unroll
        for (int ti = 0; ti < 2; ++ti) {
#pragma unroll
            for (int r = 0; r < 8; ++r) {
                const int mLoc = wm * 32 + ti * 16 + g * 8 + r;
                unsafeAtomicAdd(&sRowD[mLoc], rowE[ti][r]);
                unsafeAtomicAdd(&sRowN[mLoc], rowEW[ti][r]);
            }
        }
    }
    __syncthreads();

    // Flush block partials to global accumulators.
    if (t < BT) {
        unsafeAtomicAdd(&rowD[bm + t], sRowD[t]);
        unsafeAtomicAdd(&rowN[bm + t], sRowN[t]);
    } else {
        const int u = t - BT;
        unsafeAtomicAdd(&colD[bn + u], sColD[u]);
        unsafeAtomicAdd(&colN[bn + u], sColN[u]);
    }
}

// ---------------------------------------------------------------------------
// Kernel 4: final scalar loss from the 4 reduction vectors.
// ---------------------------------------------------------------------------
__global__ void __launch_bounds__(256)
loss_kernel(const float* __restrict__ rowD, const float* __restrict__ rowN,
            const float* __restrict__ colD, const float* __restrict__ colN,
            float* __restrict__ out) {
    const int t = threadIdx.x;
    float sa = 0.0f, ca = 0.0f, sv = 0.0f, cv = 0.0f;
    for (int i = t; i < Mm; i += 256) {
        const float na = rowN[i], da = rowD[i];
        if (na > 0.0f) { sa += __logf((na + EPS_F) / (da + EPS_F)); ca += 1.0f; }
        const float nv = colN[i], dv = colD[i];
        if (nv > 0.0f) { sv += __logf((nv + EPS_F) / (dv + EPS_F)); cv += 1.0f; }
    }
#pragma unroll
    for (int m = 16; m >= 1; m >>= 1) {
        sa += __shfl_xor(sa, m, 32);
        ca += __shfl_xor(ca, m, 32);
        sv += __shfl_xor(sv, m, 32);
        cv += __shfl_xor(cv, m, 32);
    }
    __shared__ float red[4][8];
    const int wave = t >> 5, lane = t & 31;
    if (lane == 0) { red[0][wave] = sa; red[1][wave] = ca; red[2][wave] = sv; red[3][wave] = cv; }
    __syncthreads();
    if (t == 0) {
        float Sa = 0, Ca = 0, Sv = 0, Cv = 0;
        for (int i = 0; i < 8; ++i) { Sa += red[0][i]; Ca += red[1][i]; Sv += red[2][i]; Cv += red[3][i]; }
        const float lossA = -Sa / fmaxf(Ca, 1.0f);
        const float lossV = -Sv / fmaxf(Cv, 1.0f);
        out[0] = 0.5f * (lossA + lossV);
    }
}

// ---------------------------------------------------------------------------
extern "C" void kernel_launch(void* const* d_in, const int* in_sizes, int n_in,
                              void* d_out, int out_size, void* d_ws, size_t ws_size,
                              hipStream_t stream) {
    const float* a  = (const float*)d_in[0];
    const float* v  = (const float*)d_in[1];
    const int*   la = (const int*)d_in[2];
    const int*   lv = (const int*)d_in[3];

    // Workspace layout: af (M*D f16) | vf (M*D f16) | 4*M floats of accumulators
    _Float16* af = (_Float16*)d_ws;
    _Float16* vf = af + (size_t)Mm * Dk;
    float* acc   = (float*)(vf + (size_t)Mm * Dk);
    float* rowD  = acc;
    float* rowN  = acc + Mm;
    float* colD  = acc + 2 * Mm;
    float* colN  = acc + 3 * Mm;

    // 1) normalize both feature sets into f16 (one wave per row)
    normalize_kernel<<<(2 * Mm) / 8, 256, 0, stream>>>(a, v, af, vf);

    // 2) zero global accumulators
    zero_kernel<<<(4 * Mm + 255) / 256, 256, 0, stream>>>(acc, 4 * Mm);

    // 3) fused WMMA sim tiles + exp + masked row/col reductions
    dim3 grid(Mm / BT, Mm / BT);
    sim_tile_kernel<<<grid, 256, 0, stream>>>(af, vf, la, lv, rowD, rowN, colD, colN);

    // 4) scalar loss
    loss_kernel<<<1, 256, 0, stream>>>(rowD, rowN, colD, colN, (float*)d_out);
}
